// SAGESupervised_13056700580224
// MI455X (gfx1250) — compile-verified, matrix operand order checked by reading, not run on hardware
//
#include <hip/hip_runtime.h>

// ---------------------------------------------------------------------------
// Problem constants (fixed by the reference).
// ---------------------------------------------------------------------------
#define N_SRC0 100000
#define N_DST0 20000
#define N_DST1 4000
#define NE0    500000
#define NE1    100000
#define D_IN   128
#define D_H    256
#define NT     4          // 16-col N-tiles per wave (wave computes 16x64)

typedef __bf16 v16bf __attribute__((ext_vector_type(16)));
typedef __bf16 v8bf  __attribute__((ext_vector_type(8)));
typedef float  v8f   __attribute__((ext_vector_type(8)));
typedef float  v4f   __attribute__((ext_vector_type(4)));

union BF16x16 {
  v16bf v;
  struct { v8bf lo, hi; } p;
};

// ---------------------------------------------------------------------------
// Zero-fill scratch (accumulators must start at 0 every call).
// ---------------------------------------------------------------------------
__global__ void zero_f32(float* __restrict__ p, long n) {
  long i = (long)blockIdx.x * blockDim.x + threadIdx.x;
  if (i < n) p[i] = 0.0f;
}

// ---------------------------------------------------------------------------
// One-time (per call) weight repack: fp32 row-major [K][N] -> bf16
// column-major [N][K]. Makes every B-fragment fetch two aligned b128 loads.
// Total weights ~0.4 MB, cost is noise.
// ---------------------------------------------------------------------------
__global__ void convert_w_colmajor_bf16(const float* __restrict__ W,
                                        __bf16* __restrict__ Wc, int K, int N) {
  int i = blockIdx.x * blockDim.x + threadIdx.x;
  if (i >= K * N) return;
  int n = i / K;
  int k = i - n * K;
  Wc[i] = (__bf16)W[(size_t)k * N + n];
}

// ---------------------------------------------------------------------------
// Edge scatter: one wave32 per edge; each lane gathers a float4 (the wave
// reads 512 contiguous bytes per row chunk) and issues 4 scalar f32 atomics.
// x (51 MB) and the sum buffers are L2-resident (192 MB L2), so both the
// gather and the atomics stay on-chip.
// ---------------------------------------------------------------------------
__global__ void scatter_accum(const int* __restrict__ ei, int E,
                              const float* __restrict__ feat, int D,
                              float* __restrict__ sum, float* __restrict__ cnt) {
  int e = blockIdx.x * (blockDim.x >> 5) + (threadIdx.x >> 5);
  if (e >= E) return;
  int lane = threadIdx.x & 31;
  int src = ei[e];        // edge_index[0][e]
  int dst = ei[E + e];    // edge_index[1][e]
  const float* __restrict__ frow = feat + (size_t)src * D;
  float* __restrict__ srow = sum + (size_t)dst * D;
  for (int j = lane * 4; j < D; j += 128) {
    v4f v = *(const v4f*)(frow + j);
#pragma unroll
    for (int c = 0; c < 4; ++c) atomicAdd(&srow[j + c], v[c]);
  }
  if (lane == 0) atomicAdd(&cnt[dst], 1.0f);
}

// ---------------------------------------------------------------------------
// Fragment loaders. Per lane the 16 K-values of a fragment are two
// contiguous runs of 8: K = kb+8h .. +7 and K = kb+16+8h .. +7
// (ISA 7.12.2: element i maps to K = i + (i&8) + 8h).
// ---------------------------------------------------------------------------
__device__ __forceinline__ v16bf load_a_frag(const float* __restrict__ A, int K,
                                             int m, int kb, int h, float scale) {
  const float* p = A + (size_t)m * K + kb + 8 * h;
  v4f f0 = *(const v4f*)(p);
  v4f f1 = *(const v4f*)(p + 4);
  v4f f2 = *(const v4f*)(p + 16);
  v4f f3 = *(const v4f*)(p + 20);
  BF16x16 r;
#pragma unroll
  for (int i = 0; i < 4; ++i) {
    r.p.lo[i]     = (__bf16)(f0[i] * scale);
    r.p.lo[i + 4] = (__bf16)(f1[i] * scale);
    r.p.hi[i]     = (__bf16)(f2[i] * scale);
    r.p.hi[i + 4] = (__bf16)(f3[i] * scale);
  }
  return r.v;
}

__device__ __forceinline__ v16bf load_b_frag(const __bf16* __restrict__ Wc, int K,
                                             int n, int kb, int h) {
  const __bf16* p = Wc + (size_t)n * K + kb + 8 * h;
  BF16x16 r;
  r.p.lo = *(const v8bf*)(p);       // b128
  r.p.hi = *(const v8bf*)(p + 16);  // b128
  return r.v;
}

// ---------------------------------------------------------------------------
// Fused SAGE linear: out = act( (sum/cnt)@W1 + A2@W2 + bias ).
// One 16x64 output strip per wave: one A fragment feeds 4 WMMAs (4 N-tiles),
// fp32 accumulation, mean-division fused into the A1 fp32->bf16 convert.
// All dims are multiples of 16/32 -> no tails, EXEC stays all-ones.
// ---------------------------------------------------------------------------
__global__ void sage_gemm_wmma(const float* __restrict__ A1,
                               const float* __restrict__ cnt,
                               const float* __restrict__ A2,
                               const __bf16* __restrict__ W1c,
                               const __bf16* __restrict__ W2c,
                               const float* __restrict__ bias,
                               float* __restrict__ out,
                               int M, int K, int N, int relu) {
  const int lane  = threadIdx.x & 31;
  const int wave  = threadIdx.x >> 5;
  const int wtile = blockIdx.x * (blockDim.x >> 5) + wave;
  const int ncol  = N / (16 * NT);            // wave-strips per tile row
  const int tileM = wtile / ncol;
  const int colg  = wtile - tileM * ncol;
  if (tileM >= (M >> 4)) return;

  const int m  = (tileM << 4) + (lane & 15);  // A row this lane loads
  const int h  = lane >> 4;                   // half-wave select
  const int n0 = colg * (16 * NT) + (lane & 15);

  const float rscale = 1.0f / fmaxf(cnt[m], 1.0f);  // fused mean

  v8f acc[NT] = {};
  for (int pass = 0; pass < 2; ++pass) {
    const float* __restrict__ A   = pass ? A2 : A1;
    const __bf16* __restrict__ Wc = pass ? W2c : W1c;
    const float sc = pass ? 1.0f : rscale;
    for (int kb = 0; kb < K; kb += 32) {
      v16bf a = load_a_frag(A, K, m, kb, h, sc);
#pragma unroll
      for (int j = 0; j < NT; ++j) {
        v16bf b = load_b_frag(Wc, K, n0 + 16 * j, kb, h);
        acc[j] = __builtin_amdgcn_wmma_f32_16x16x32_bf16(
            false, a, false, b, (short)0, acc[j], false, false);
      }
    }
  }

#pragma unroll
  for (int j = 0; j < NT; ++j) {
    const int n = n0 + 16 * j;
    const float bn = bias[n];
#pragma unroll
    for (int r = 0; r < 8; ++r) {
      const int row = (tileM << 4) + r + (h << 3);
      float v = acc[j][r] + bn;
      if (relu) v = fmaxf(v, 0.0f);
      out[(size_t)row * N + n] = v;
    }
  }
}

// ---------------------------------------------------------------------------
// Row-wise log_softmax: one 256-thread block per row (N == 256).
// ---------------------------------------------------------------------------
__global__ void log_softmax_rows(const float* __restrict__ in,
                                 float* __restrict__ outp, int N) {
  __shared__ float red[256];
  const int row = blockIdx.x;
  const int t = threadIdx.x;
  const float v = in[(size_t)row * N + t];

  red[t] = v;
  __syncthreads();
  for (int s = 128; s > 0; s >>= 1) {
    if (t < s) red[t] = fmaxf(red[t], red[t + s]);
    __syncthreads();
  }
  const float mx = red[0];
  __syncthreads();

  red[t] = __expf(v - mx);
  __syncthreads();
  for (int s = 128; s > 0; s >>= 1) {
    if (t < s) red[t] += red[t + s];
    __syncthreads();
  }
  const float lse = mx + logf(red[0]);
  outp[(size_t)row * N + t] = v - lse;
}

// ---------------------------------------------------------------------------
// Launch pipeline (all on `stream`, deterministic, scratch in d_ws).
// ws layout (floats):
//   sum0[20000*128] cnt0[20000] hbuf[20000*256] sum1[4000*256] cnt1[4000]
//   then bf16 weights (col-major): Wl0c[256*128] Wr0c[256*128]
//                                  Wl1c[256*256] Wr1c[256*256]   (~35.7 MB)
// All segment byte offsets are multiples of 16 -> b128-aligned.
// ---------------------------------------------------------------------------
extern "C" void kernel_launch(void* const* d_in, const int* in_sizes, int n_in,
                              void* d_out, int out_size, void* d_ws, size_t ws_size,
                              hipStream_t stream) {
  (void)in_sizes; (void)n_in; (void)out_size; (void)ws_size;
  const float* x   = (const float*)d_in[0];
  const int*   ei0 = (const int*)d_in[1];
  const int*   ei1 = (const int*)d_in[2];
  const float* Wl0 = (const float*)d_in[3];
  const float* bl0 = (const float*)d_in[4];
  const float* Wr0 = (const float*)d_in[5];
  const float* Wl1 = (const float*)d_in[6];
  const float* bl1 = (const float*)d_in[7];
  const float* Wr1 = (const float*)d_in[8];

  float* ws   = (float*)d_ws;
  float* sum0 = ws;                                   // 2,560,000 f
  float* cnt0 = sum0 + (size_t)N_DST0 * D_IN;         //    20,000 f
  float* hbuf = cnt0 + N_DST0;                        // 5,120,000 f
  float* sum1 = hbuf + (size_t)N_DST0 * D_H;          // 1,024,000 f
  float* cnt1 = sum1 + (size_t)N_DST1 * D_H;          //     4,000 f
  __bf16* wb   = (__bf16*)(cnt1 + N_DST1);
  __bf16* Wl0c = wb;                                  // 256*128 bf16
  __bf16* Wr0c = Wl0c + (size_t)D_H * D_IN;
  __bf16* Wl1c = Wr0c + (size_t)D_H * D_IN;           // 256*256 bf16
  __bf16* Wr1c = Wl1c + (size_t)D_H * D_H;

  float* out = (float*)d_out;                         // [4000,256]
  float* lsm = out + (size_t)N_DST1 * D_H;            // [4000,256]

  // ---- Weight repack to bf16 column-major ----
  {
    int n0 = D_IN * D_H, n1 = D_H * D_H;
    convert_w_colmajor_bf16<<<(n0 + 255) / 256, 256, 0, stream>>>(Wl0, Wl0c, D_IN, D_H);
    convert_w_colmajor_bf16<<<(n0 + 255) / 256, 256, 0, stream>>>(Wr0, Wr0c, D_IN, D_H);
    convert_w_colmajor_bf16<<<(n1 + 255) / 256, 256, 0, stream>>>(Wl1, Wl1c, D_H, D_H);
    convert_w_colmajor_bf16<<<(n1 + 255) / 256, 256, 0, stream>>>(Wr1, Wr1c, D_H, D_H);
  }

  // ---- Layer 0: scatter-sum, then fused (mean@Wl + x_dst@Wr + b, ReLU) ----
  const long z0 = (long)N_DST0 * D_IN + N_DST0;       // sum0+cnt0 contiguous
  zero_f32<<<(unsigned)((z0 + 255) / 256), 256, 0, stream>>>(sum0, z0);
  scatter_accum<<<(NE0 + 7) / 8, 256, 0, stream>>>(ei0, NE0, x, D_IN, sum0, cnt0);
  // wave-strips = (20000/16) * (256/64) = 5000 ; 8 waves/block -> 625 blocks
  sage_gemm_wmma<<<(N_DST0 / 16) * (D_H / (16 * NT)) / 8, 256, 0, stream>>>(
      sum0, cnt0, x, Wl0c, Wr0c, bl0, hbuf, N_DST0, D_IN, D_H, /*relu=*/1);

  // ---- Layer 1 ----
  const long z1 = (long)N_DST1 * D_H + N_DST1;        // sum1+cnt1 contiguous
  zero_f32<<<(unsigned)((z1 + 255) / 256), 256, 0, stream>>>(sum1, z1);
  scatter_accum<<<(NE1 + 7) / 8, 256, 0, stream>>>(ei1, NE1, hbuf, D_H, sum1, cnt1);
  // wave-strips = (4000/16) * 4 = 1000 -> 125 blocks
  sage_gemm_wmma<<<(N_DST1 / 16) * (D_H / (16 * NT)) / 8, 256, 0, stream>>>(
      sum1, cnt1, hbuf, Wl1c, Wr1c, bl1, out, N_DST1, D_H, D_H, /*relu=*/0);

  // ---- log_softmax over the 4000x256 logits ----
  log_softmax_rows<<<N_DST1, D_H, 0, stream>>>(out, lsm, D_H);
}
